// OptimizedGatedMultiHeadAttention_47991964565622
// MI455X (gfx1250) — matrix operand order repaired and checked by reading, not compiled, not run on hardware
//
#include <hip/hip_runtime.h>
#include <cstdint>

#define B_  2
#define S_  2048
#define H_  1024
#define NH_ 16
#define HD_ 64
#define N_TOK (B_ * S_)
#define H3_ (3 * H_)

typedef __attribute__((ext_vector_type(16))) __bf16 v16bf;
typedef __attribute__((ext_vector_type(8)))  float  v8f;
typedef __attribute__((ext_vector_type(4)))  uint32_t u32x4;
typedef __attribute__((ext_vector_type(8)))  uint32_t u32x8;

union FragU { v16bf v; uint32_t u[8]; };

__device__ __forceinline__ v8f v8f_zero() {
  v8f z;
#pragma unroll
  for (int i = 0; i < 8; ++i) z[i] = 0.0f;
  return z;
}

// 16-bit A fragment, 16x32 (MxK): lane holds row M = lane%16.
// dword d (0..7): g=d/4, q=d%4 -> K = g*16 + (lane>=16 ? 8 : 0) + 2q (pair)
__device__ __forceinline__ v16bf frag_a(const uint32_t* rowp, int hi) {
  FragU f;
#pragma unroll
  for (int g = 0; g < 2; ++g)
#pragma unroll
    for (int q = 0; q < 4; ++q)
      f.u[g * 4 + q] = rowp[g * 8 + hi * 4 + q];
  return f.v;
}

// 16-bit B fragment, 32x16 (KxN): lane holds column N = lane%16.
// dword j: g=j/4, q=j%4 -> K = g*16 + (lane>=16 ? 8 : 0) + 2q (pair)
__device__ __forceinline__ v16bf frag_b(const uint32_t* rowp, int hi) {
  FragU f;
#pragma unroll
  for (int j = 0; j < 8; ++j) f.u[j] = rowp[(j >> 2) * 8 + hi * 4 + (j & 3)];
  return f.v;
}

__device__ __forceinline__ v8f wmma_bf16(v16bf a, v16bf b, v8f c) {
  // (neg_a, A, neg_b, B, c_mod, C, reuse_a, reuse_b)
  return __builtin_amdgcn_wmma_f32_16x16x32_bf16(false, a, false, b, (short)0, c,
                                                 false, false);
}

__device__ __forceinline__ uint32_t bf16_bits(float x) {
  union { __bf16 h; unsigned short s; } u;
  u.h = (__bf16)x;
  return (uint32_t)u.s;
}

// ------------------------------------------------------------ TDM helpers
// D# group0: count=1 (valid user descriptor), LDS address, 57-bit global
// byte address, type=2 ("image").
__device__ __forceinline__ u32x4 tdm_g0(uint32_t lds_addr, uint64_t gaddr) {
  u32x4 g;
  g[0] = 1u;
  g[1] = lds_addr;
  g[2] = (uint32_t)gaddr;
  g[3] = ((uint32_t)(gaddr >> 32) & 0x01FFFFFFu) | (2u << 30);
  return g;
}

// D# group1 for a 2D bf16 tile: data_size=2B, LDS padding enabled.
// pad_interval: 0=2,1=4,2=8,3=16,4=32,... dwords ; pad_amount: value+1 dwords
__device__ __forceinline__ u32x8 tdm_g1_2d(uint32_t tile_w, uint32_t tile_h,
                                           uint32_t tens_w, uint32_t tens_h,
                                           uint32_t row_stride,
                                           uint32_t pad_interval,
                                           uint32_t pad_amount) {
  u32x8 g;
  g[0] = (1u << 16) | (1u << 20) | (pad_interval << 22) | (pad_amount << 25);
  g[1] = (tens_w & 0xFFFFu) << 16;                                  // tensor_dim0 lo
  g[2] = ((tens_w >> 16) & 0xFFFFu) | ((tens_h & 0xFFFFu) << 16);   // dim0 hi | dim1 lo
  g[3] = ((tens_h >> 16) & 0xFFFFu) | ((tile_w & 0xFFFFu) << 16);   // dim1 hi | tile_dim0
  g[4] = (tile_h & 0xFFFFu);                                        // tile_dim1, tile_dim2=0
  g[5] = row_stride;                                                // tensor_dim0_stride lo
  g[6] = 0;                                                         // stride hi | dim1_stride lo
  g[7] = 0;
  return g;
}

__device__ __forceinline__ void tdm_load_2d(u32x4 g0, u32x8 g1) {
  asm volatile("tensor_load_to_lds %0, %1" : : "s"(g0), "s"(g1) : "memory");
}

__device__ __forceinline__ uint32_t lds_addr_of(const void* p) {
  // generic pointers to LDS carry the wave-relative LDS byte address in the
  // low 32 bits
  return (uint32_t)(uintptr_t)p;
}

// ---------------------------------------------------------------- convert
__global__ void cvt_f32_to_bf16(const float* __restrict__ in,
                                __bf16* __restrict__ out, int n4) {
  int i = blockIdx.x * blockDim.x + threadIdx.x;
  if (i >= n4) return;
  float4 v = reinterpret_cast<const float4*>(in)[i];
  union { __bf16 h[4]; uint2 u; } r;
  r.h[0] = (__bf16)v.x; r.h[1] = (__bf16)v.y;
  r.h[2] = (__bf16)v.z; r.h[3] = (__bf16)v.w;
  reinterpret_cast<uint2*>(out)[i] = r.u;
}

// ------------------------------------------------- NT GEMM: C = A * B^T
// A: [M,K] bf16 row-major, Bm: [N,K] bf16 row-major, C: [M,N]
template <bool OUT_BF16>
__global__ __launch_bounds__(256) void gemm_nt_bf16(
    const __bf16* __restrict__ A, const __bf16* __restrict__ Bm,
    void* __restrict__ Cout, int M, int N, int K) {
  __shared__ uint32_t As[128][24];  // 128 rows x 32 k (bf16) + 16 bf16 pad
  __shared__ uint32_t Bs[128][24];
  const int tid = threadIdx.x;
  const int wave = tid >> 5, lane = tid & 31;
  const int lane16 = lane & 15, hi = lane >> 4;
  const int wm = wave >> 1, wn = wave & 1;  // 4 x 2 wave grid
  const int m0 = blockIdx.y * 128, n0 = blockIdx.x * 128;

  const uint32_t ldsA = lds_addr_of(&As[0][0]);
  const uint32_t ldsB = lds_addr_of(&Bs[0][0]);
  // 128x32 bf16 tile: row = 16 dwords, pad 8 dwords -> interval idx 3, amount 7
  const u32x8 g1 = tdm_g1_2d(32u, 128u, 32u, 128u, (uint32_t)K, 3u, 7u);

  v8f acc[2][4];
#pragma unroll
  for (int a = 0; a < 2; ++a)
#pragma unroll
    for (int b = 0; b < 4; ++b) acc[a][b] = v8f_zero();

  for (int kb = 0; kb < K; kb += 32) {
    if (wave == 0) {
      tdm_load_2d(tdm_g0(ldsA, (uint64_t)(uintptr_t)(A + (size_t)m0 * K + kb)), g1);
      tdm_load_2d(tdm_g0(ldsB, (uint64_t)(uintptr_t)(Bm + (size_t)n0 * K + kb)), g1);
      __builtin_amdgcn_s_wait_tensorcnt(0);
    }
    __syncthreads();

    v16bf af[2], bfv[4];
#pragma unroll
    for (int a = 0; a < 2; ++a)
      af[a] = frag_a(&As[wm * 32 + a * 16 + lane16][0], hi);
#pragma unroll
    for (int b = 0; b < 4; ++b)
      bfv[b] = frag_b(&Bs[wn * 64 + b * 16 + lane16][0], hi);
#pragma unroll
    for (int a = 0; a < 2; ++a)
#pragma unroll
      for (int b = 0; b < 4; ++b)
        acc[a][b] = wmma_bf16(af[a], bfv[b], acc[a][b]);
    __syncthreads();
  }

#pragma unroll
  for (int a = 0; a < 2; ++a) {
    int mrow = m0 + wm * 32 + a * 16 + hi * 8;
#pragma unroll
    for (int b = 0; b < 4; ++b) {
      int col = n0 + wn * 64 + b * 16 + lane16;
#pragma unroll
      for (int j = 0; j < 8; ++j) {
        float v = acc[a][b][j];
        size_t idx = (size_t)(mrow + j) * N + col;
        if (OUT_BF16) reinterpret_cast<__bf16*>(Cout)[idx] = (__bf16)v;
        else          reinterpret_cast<float*>(Cout)[idx] = v;
      }
    }
  }
}

// --------------------------------------------------- flash attention
// qkv: [N_TOK, 3*H] bf16 ([b*s, {q,k,v} x NH x HD]); ctxb: [N_TOK, H] bf16
__global__ __launch_bounds__(256) void attn_flash(
    const __bf16* __restrict__ qkv, const float* __restrict__ mask,
    const float* __restrict__ gate, __bf16* __restrict__ ctxb) {
  __shared__ uint32_t Ks[128][36];      // 128 keys x 64 d (bf16) + 8 bf16 pad
  __shared__ uint32_t Vt[64][66];       // transposed V: [d][key-pair packed]
  __shared__ uint32_t Ps[8][16][34];    // per-wave P half-tile: 16 x 64 keys

  const int tid = threadIdx.x;
  const int wave = tid >> 5, lane = tid & 31;
  const int lane16 = lane & 15, hi = lane >> 4;
  const int qb = blockIdx.x * 128;
  const int bh = blockIdx.y;
  const int b = bh >> 4, h = bh & 15;

  const uint32_t ldsKs = lds_addr_of(&Ks[0][0]);
  // 128x64 bf16 tile: row = 32 dwords, pad 4 dwords -> interval idx 4, amount 3
  const u32x8 g1K = tdm_g1_2d(64u, 128u, 64u, 128u, (uint32_t)H3_, 4u, 3u);

  // Q fragments for this wave's 16 query rows, kept in registers all kernel
  const uint32_t* qkv32 = reinterpret_cast<const uint32_t*>(qkv);
  const int qrow = qb + wave * 16 + lane16;
  const size_t qbase32 = (((size_t)(b * S_ + qrow)) * H3_ + h * HD_) >> 1;
  const v16bf qa0 = frag_a(qkv32 + qbase32, hi);       // k = 0..31
  const v16bf qa1 = frag_a(qkv32 + qbase32 + 16, hi);  // k = 32..63

  float mrow[8], lrow[8], alpha[8];
  v8f accO[4];
#pragma unroll
  for (int j = 0; j < 8; ++j) { mrow[j] = -3.0e38f; lrow[j] = 0.0f; }
#pragma unroll
  for (int d = 0; d < 4; ++d) accO[d] = v8f_zero();

  const float LOG2E = 1.4426950408889634f;

  for (int kv0 = 0; kv0 < S_; kv0 += 128) {
    const size_t kBase = ((size_t)(b * S_ + kv0)) * H3_ + H_ + h * HD_;
    const size_t vBase = ((size_t)(b * S_ + kv0)) * H3_ + 2 * H_ + h * HD_;

    // K tile via Tensor Data Mover (wave 0 issues; TDM ignores EXEC)
    if (wave == 0)
      tdm_load_2d(tdm_g0(ldsKs, (uint64_t)(uintptr_t)(qkv + kBase)), g1K);

#pragma unroll
    for (int i = 0; i < 2; ++i) {           // V tile, transposed + pair-packed
      int s = tid + i * 256;
      int rp = s >> 3, dd = (s & 7) * 8;    // rp = key pair, dd = d base
      uint4 va = *reinterpret_cast<const uint4*>(qkv + vBase + (size_t)(2 * rp) * H3_ + dd);
      uint4 vb = *reinterpret_cast<const uint4*>(qkv + vBase + (size_t)(2 * rp + 1) * H3_ + dd);
      const unsigned short* pa = reinterpret_cast<const unsigned short*>(&va);
      const unsigned short* pb = reinterpret_cast<const unsigned short*>(&vb);
#pragma unroll
      for (int e = 0; e < 8; ++e)
        Vt[dd + e][rp] = (uint32_t)pa[e] | ((uint32_t)pb[e] << 16);
    }
    if (wave == 0) __builtin_amdgcn_s_wait_tensorcnt(0);
    __syncthreads();

    // ---- scores: 16 queries x 128 keys, scaled + mask
    v8f st[8];
#pragma unroll
    for (int t = 0; t < 8; ++t) {
      float mval = mask[(size_t)b * S_ + kv0 + t * 16 + lane16];
      v16bf kb0 = frag_b(&Ks[t * 16 + lane16][0], hi);
      v16bf kb1 = frag_b(&Ks[t * 16 + lane16][16], hi);
      v8f s0 = wmma_bf16(qa0, kb0, v8f_zero());
      s0 = wmma_bf16(qa1, kb1, s0);
#pragma unroll
      for (int j = 0; j < 8; ++j) st[t][j] = s0[j] * 0.125f + mval;
    }

    // ---- online softmax (rows live in (VGPR j, lane-half))
#pragma unroll
    for (int j = 0; j < 8; ++j) {
      float tm = st[0][j];
#pragma unroll
      for (int t = 1; t < 8; ++t) tm = fmaxf(tm, st[t][j]);
      tm = fmaxf(tm, __shfl_xor(tm, 1, 32));
      tm = fmaxf(tm, __shfl_xor(tm, 2, 32));
      tm = fmaxf(tm, __shfl_xor(tm, 4, 32));
      tm = fmaxf(tm, __shfl_xor(tm, 8, 32));
      float mnew = fmaxf(mrow[j], tm);
      alpha[j] = exp2f((mrow[j] - mnew) * LOG2E);
      mrow[j] = mnew;
      float rs = 0.0f;
#pragma unroll
      for (int t = 0; t < 8; ++t) {
        float p = exp2f((st[t][j] - mnew) * LOG2E);
        st[t][j] = p;
        rs += p;
      }
      rs += __shfl_xor(rs, 1, 32);
      rs += __shfl_xor(rs, 2, 32);
      rs += __shfl_xor(rs, 4, 32);
      rs += __shfl_xor(rs, 8, 32);
      lrow[j] = lrow[j] * alpha[j] + rs;
    }
#pragma unroll
    for (int d = 0; d < 4; ++d)
#pragma unroll
      for (int j = 0; j < 8; ++j) accO[d][j] *= alpha[j];

    // ---- P @ V, 64 keys per half (P restaged via per-wave LDS)
#pragma unroll
    for (int h2 = 0; h2 < 2; ++h2) {
#pragma unroll
      for (int tt = 0; tt < 4; ++tt) {
        int t = h2 * 4 + tt;
#pragma unroll
        for (int j = 0; j < 8; ++j) {
          float p = st[t][j];
          float pq = __shfl_xor(p, 1, 32);
          if ((lane & 1) == 0) {
            uint32_t w = bf16_bits(p) | (bf16_bits(pq) << 16);
            Ps[wave][hi * 8 + j][tt * 8 + (lane16 >> 1)] = w;
          }
        }
      }
      v16bf pa0 = frag_a(&Ps[wave][lane16][0], hi);   // keys rel 0..31
      v16bf pa1 = frag_a(&Ps[wave][lane16][16], hi);  // keys rel 32..63
#pragma unroll
      for (int d = 0; d < 4; ++d) {
        v16bf vb0 = frag_b(&Vt[d * 16 + lane16][h2 * 32], hi);
        v16bf vb1 = frag_b(&Vt[d * 16 + lane16][h2 * 32 + 16], hi);
        accO[d] = wmma_bf16(pa0, vb0, accO[d]);
        accO[d] = wmma_bf16(pa1, vb1, accO[d]);
      }
    }
    __syncthreads();
  }

  const float gv = gate[h];
  float inv[8];
#pragma unroll
  for (int j = 0; j < 8; ++j) inv[j] = gv / lrow[j];
#pragma unroll
  for (int d = 0; d < 4; ++d)
#pragma unroll
    for (int j = 0; j < 8; ++j) {
      int r = qb + wave * 16 + hi * 8 + j;
      int col = h * HD_ + d * 16 + lane16;
      ctxb[((size_t)(b * S_ + r)) * H_ + col] = (__bf16)(accO[d][j] * inv[j]);
    }
}

// ----------------------------------------------------------------- launcher
extern "C" void kernel_launch(void* const* d_in, const int* in_sizes, int n_in,
                              void* d_out, int out_size, void* d_ws, size_t ws_size,
                              hipStream_t stream) {
  const float* hs   = (const float*)d_in[0];
  const float* mask = (const float*)d_in[1];
  const float* wqkv = (const float*)d_in[2];
  const float* wout = (const float*)d_in[3];
  const float* gate = (const float*)d_in[4];
  float* out = (float*)d_out;

  char* w = (char*)d_ws;
  size_t off = 0;
  auto wsalloc = [&](size_t bytes) -> void* {
    void* p = (void*)(w + off);
    off = (off + bytes + 255) & ~(size_t)255;
    return p;
  };
  __bf16* Xb    = (__bf16*)wsalloc((size_t)N_TOK * H_ * 2);   // bf16 hidden
  __bf16* Wqkvb = (__bf16*)wsalloc((size_t)H3_ * H_ * 2);     // bf16 w_qkv
  __bf16* Woutb = (__bf16*)wsalloc((size_t)H_ * H_ * 2);      // bf16 w_out
  __bf16* qkvb  = (__bf16*)wsalloc((size_t)N_TOK * H3_ * 2);  // bf16 qkv
  __bf16* ctxb  = (__bf16*)wsalloc((size_t)N_TOK * H_ * 2);   // bf16 context
  (void)ws_size; (void)in_sizes; (void)n_in; (void)out_size;

  {
    int n4 = N_TOK * H_ / 4;
    cvt_f32_to_bf16<<<(n4 + 255) / 256, 256, 0, stream>>>(hs, Xb, n4);
  }
  {
    int n4 = H3_ * H_ / 4;
    cvt_f32_to_bf16<<<(n4 + 255) / 256, 256, 0, stream>>>(wqkv, Wqkvb, n4);
  }
  {
    int n4 = H_ * H_ / 4;
    cvt_f32_to_bf16<<<(n4 + 255) / 256, 256, 0, stream>>>(wout, Woutb, n4);
  }

  // qkv = X @ Wqkv^T   [4096 x 3072]
  gemm_nt_bf16<true><<<dim3(H3_ / 128, N_TOK / 128), 256, 0, stream>>>(
      Xb, Wqkvb, (void*)qkvb, N_TOK, H3_, H_);

  // fused flash attention + gate -> ctx bf16 [4096 x 1024]
  attn_flash<<<dim3(S_ / 128, B_ * NH_), 256, 0, stream>>>(qkvb, mask, gate, ctxb);

  // out = ctx @ Wout^T  [4096 x 1024] fp32
  gemm_nt_bf16<false><<<dim3(H_ / 128, N_TOK / 128), 256, 0, stream>>>(
      ctxb, Woutb, (void*)out, N_TOK, H_, H_);
}
